// TransitionDown_71064528879924
// MI455X (gfx1250) — matrix operand order repaired and checked by reading, not compiled, not run on hardware
//
#include <hip/hip_runtime.h>
#include <stdint.h>

// ---------------------------------------------------------------------------
// Problem constants (match reference)
// ---------------------------------------------------------------------------
#define B_   16
#define N_   4096
#define D_   254
#define S_   1024            // npoint = N/4
#define K_   16              // NSAMPLE
#define M_   (B_ * S_ * K_)  // 262144 rows through the MLP
#define C1IN 256             // 2 + D
#define COUT 512

typedef __bf16 v16bf __attribute__((ext_vector_type(16)));
typedef float  v8f   __attribute__((ext_vector_type(8)));

__device__ __forceinline__ float bf2f(unsigned short h) {
  return __uint_as_float(((unsigned)h) << 16);
}
__device__ __forceinline__ unsigned short f2bf(float f) {
  unsigned u = __float_as_uint(f);
  u += 0x7FFFu + ((u >> 16) & 1u);   // round-to-nearest-even
  return (unsigned short)(u >> 16);
}

// CDNA5 async global->LDS copy (16B per lane), tracked by ASYNCcnt.
// GVS addressing: mem_addr = SGPR_u64(base) + VGPR_i32(byte offset).
// LDS operand is the wave-relative LDS byte offset = low 32 bits of the flat
// shared-pointer address (ISA 10.2).
__device__ __forceinline__ void async_ld16_s(void* lds, const void* base,
                                             unsigned voff) {
  asm volatile("global_load_async_to_lds_b128 %0, %1, %2"
               :: "v"((unsigned)(unsigned long long)lds), "v"(voff), "s"(base)
               : "memory");
}
__device__ __forceinline__ void wait_async() {
  asm volatile("s_wait_asynccnt 0" ::: "memory");
}

// ---------------------------------------------------------------------------
// 1) Farthest point sampling: one 1024-thread block per batch.
//    u64 key = (dist_bits << 32) | ~idx  -> max-reduce == argmax, ties -> low idx
// ---------------------------------------------------------------------------
__global__ __launch_bounds__(1024) void fps_kernel(const float* __restrict__ xy,
                                                   int* __restrict__ fps_idx,
                                                   float* __restrict__ newxy) {
  const int b = blockIdx.x, tid = threadIdx.x;
  const int lane = tid & 31, wave = tid >> 5;
  float px[4], py[4], dmin[4];
#pragma unroll
  for (int i = 0; i < 4; i++) {
    int j = tid * 4 + i;
    px[i] = xy[((long)b * N_ + j) * 2 + 0];
    py[i] = xy[((long)b * N_ + j) * 2 + 1];
    dmin[i] = 1e10f;
  }
  __shared__ unsigned long long swav[32];
  __shared__ float scx, scy;
  __shared__ int sfar;
  if (tid == 0) sfar = 0;
  __syncthreads();
  for (int s = 0; s < S_; s++) {
    if (tid == 0) {
      int far = sfar;
      fps_idx[b * S_ + s] = far;
      float cx = xy[((long)b * N_ + far) * 2 + 0];
      float cy = xy[((long)b * N_ + far) * 2 + 1];
      scx = cx; scy = cy;
      newxy[(b * S_ + s) * 2 + 0] = cx;
      newxy[(b * S_ + s) * 2 + 1] = cy;
    }
    __syncthreads();
    float cx = scx, cy = scy;
    unsigned long long best = 0ull;
#pragma unroll
    for (int i = 0; i < 4; i++) {
      float dx = px[i] - cx, dy = py[i] - cy;
      float d = dx * dx + dy * dy;
      dmin[i] = fminf(dmin[i], d);
      unsigned long long key =
          ((unsigned long long)__float_as_uint(dmin[i]) << 32) |
          (unsigned)(~(unsigned)(tid * 4 + i));
      best = key > best ? key : best;
    }
#pragma unroll
    for (int off = 16; off > 0; off >>= 1) {
      unsigned long long o = __shfl_xor(best, off, 32);
      best = o > best ? o : best;
    }
    if (lane == 0) swav[wave] = best;
    __syncthreads();
    if (wave == 0) {
      unsigned long long v = swav[lane];
#pragma unroll
      for (int off = 16; off > 0; off >>= 1) {
        unsigned long long o = __shfl_xor(v, off, 32);
        v = o > v ? o : v;
      }
      if (lane == 0) sfar = (int)(~(unsigned)(v & 0xFFFFFFFFull));
    }
    __syncthreads();
  }
}

// ---------------------------------------------------------------------------
// 2) 16-NN: one wave per query. Per-lane sorted top-16 over N/32 points,
//    then wave-wide 16-step u64 min-merge (idx in low bits for stable ties).
// ---------------------------------------------------------------------------
__global__ __launch_bounds__(128) void knn_kernel(const float* __restrict__ xy,
                                                  const int* __restrict__ fps_idx,
                                                  int* __restrict__ knn_idx) {
  const int lane = threadIdx.x & 31, wave = threadIdx.x >> 5;
  const int q = blockIdx.x * 4 + wave;
  const int b = q >> 10;
  const int fi = fps_idx[q];
  const float qx = xy[((long)b * N_ + fi) * 2 + 0];
  const float qy = xy[((long)b * N_ + fi) * 2 + 1];
  float kd[K_]; int ki[K_];
#pragma unroll
  for (int t = 0; t < K_; t++) { kd[t] = 3.0e38f; ki[t] = 0; }
  for (int j = lane; j < N_; j += 32) {
    float dx = xy[((long)b * N_ + j) * 2 + 0] - qx;
    float dy = xy[((long)b * N_ + j) * 2 + 1] - qy;
    float d = dx * dx + dy * dy;
    if (d < kd[K_ - 1]) {
      int p = K_ - 1;
      while (p > 0 && d < kd[p - 1]) { kd[p] = kd[p - 1]; ki[p] = ki[p - 1]; p--; }
      kd[p] = d; ki[p] = j;
    }
  }
  int head = 0;
  for (int t = 0; t < K_; t++) {
    unsigned long long key =
        (head < K_)
            ? ((unsigned long long)__float_as_uint(kd[head]) << 32) | (unsigned)ki[head]
            : ~0ull;
    unsigned long long mn = key;
#pragma unroll
    for (int off = 16; off > 0; off >>= 1) {
      unsigned long long o = __shfl_xor(mn, off, 32);
      mn = o < mn ? o : mn;
    }
    if (key == mn && head < K_) head++;
    if (lane == 0) knn_idx[(long)q * K_ + t] = (int)(mn & 0xFFFFFFFFull);
  }
}

// ---------------------------------------------------------------------------
// 3) Gather + concat [xy_norm | points] -> bf16 A1 matrix [M_, 256]
// ---------------------------------------------------------------------------
__global__ __launch_bounds__(256) void build_a1_kernel(
    const float* __restrict__ xy, const float* __restrict__ points,
    const int* __restrict__ knn_idx, const float* __restrict__ newxy,
    unsigned short* __restrict__ a1) {
  long gid = (long)blockIdx.x * 256 + threadIdx.x;  // M_ * 256 elements
  int c = (int)(gid & 255);
  long r = gid >> 8;
  long q = r >> 4;
  int k = (int)(r & 15);
  int b = (int)(q >> 10);
  int j = knn_idx[q * K_ + k];
  float v;
  if (c < 2) v = xy[((long)b * N_ + j) * 2 + c] - newxy[q * 2 + c];
  else       v = points[((long)b * N_ + j) * D_ + (c - 2)];
  a1[gid] = f2bf(v);
}

// ---------------------------------------------------------------------------
// Small utilities
// ---------------------------------------------------------------------------
// Convert W [kdim, 512] f32 -> Wt [512, kdim] bf16 (one-time; takes the B-tile
// transpose out of the GEMM inner loop entirely).
__global__ void transpose_w_kernel(const float* __restrict__ w,
                                   unsigned short* __restrict__ wt, int kdim) {
  int idx = blockIdx.x * 256 + threadIdx.x;  // kdim * 512, coalesced read
  int n = idx & (COUT - 1);
  int k = idx >> 9;
  wt[(long)n * kdim + k] = f2bf(w[idx]);
}
__global__ void zero_f32_kernel(float* p, int n) {
  int i = blockIdx.x * 256 + threadIdx.x;
  if (i < n) p[i] = 0.f;
}

// ---------------------------------------------------------------------------
// 4) bf16 WMMA GEMM: Y[M,512] = A[M,KDIM] @ Wt^T   (Wt is [512, KDIM] bf16)
//    Block tile 128x128, K-chunk 64, double-buffered LDS, 8 waves (2M x 4N),
//    wave tile 64x32 (4x2 accumulators -> 8 WMMAs per 12 frag loads).
//    Pure-copy tiles use GLOBAL_LOAD_ASYNC_TO_LDS_B128 in GVS mode
//    (SGPR base + 32-bit lane offset -> no 64-bit per-lane addressing);
//    BN_A fuses x = relu(A*scale + shift) into the A-tile load (layer 2),
//    which needs the data in registers, so that path stays staged.
//    LDS rows padded to 72 elements to stagger banks for ds_load_b128.
// ---------------------------------------------------------------------------
template <int KDIM, bool BN_A>
__global__ __launch_bounds__(256) void gemm_bf16_kernel(
    const unsigned short* __restrict__ A, const unsigned short* __restrict__ Wt,
    unsigned short* __restrict__ Y, const float* __restrict__ ss) {
  constexpr int BM = 128, BN = 128, BK = 64;
  constexpr int LDT = BK + 8;  // padded LDS row stride (elements, 16B-aligned)
  __shared__ unsigned short sA[2][BM * LDT];  // 2 x 18 KB
  __shared__ unsigned short sB[2][BN * LDT];  // 2 x 18 KB
  const int tid = threadIdx.x;
  const int lane = tid & 31, wave = tid >> 5;
  const int wm = (wave & 1) * 64;   // 2 waves along M
  const int wn = (wave >> 1) * 32;  // 4 waves along N
  const long m0 = (long)blockIdx.x * BM;
  const int n0 = blockIdx.y * BN;

  // Per-thread tile offsets (bytes), invariant over the K loop.
  unsigned aoff[4], boff[4];
  int arow[4], ac8[4];
#pragma unroll
  for (int i = 0; i < 4; i++) {
    int lin = tid + i * 256;
    int row = lin >> 3;       // 8 x 16B per 64-elem row
    int c8 = (lin & 7) * 8;   // element offset within the K-chunk
    arow[i] = row; ac8[i] = c8;
    aoff[i] = (unsigned)(((m0 + row) * KDIM + c8) * 2);
    boff[i] = (unsigned)(((long)(n0 + row) * KDIM + c8) * 2);
  }

  v8f acc[4][2] = {};

  auto load_stage = [&](int k0, int buf) {
    const unsigned kb = (unsigned)(k0 * 2);
    // ---- A tile: BM x BK = 1024 x 16B, 4 per thread ----
#pragma unroll
    for (int i = 0; i < 4; i++) {
      if constexpr (BN_A) {
        union { uint4 u; unsigned short h[8]; } in, out;
        in.u = *(const uint4*)((const char*)A + aoff[i] + kb);
#pragma unroll
        for (int j = 0; j < 8; j++) {
          int ch = k0 + ac8[i] + j;
          float v = fmaf(bf2f(in.h[j]), ss[ch], ss[512 + ch]);
          out.h[j] = f2bf(fmaxf(v, 0.f));
        }
        *(uint4*)&sA[buf][arow[i] * LDT + ac8[i]] = out.u;
      } else {
        async_ld16_s(&sA[buf][arow[i] * LDT + ac8[i]], A, aoff[i] + kb);
      }
    }
    // ---- B tile: BN x BK = 1024 x 16B from pre-transposed Wt (async) ----
#pragma unroll
    for (int i = 0; i < 4; i++) {
      async_ld16_s(&sB[buf][arow[i] * LDT + ac8[i]], Wt, boff[i] + kb);
    }
  };

  load_stage(0, 0);
  wait_async();
  __syncthreads();
  int buf = 0;
  for (int k0 = 0; k0 < KDIM; k0 += BK) {
    if (k0 + BK < KDIM) load_stage(k0 + BK, buf ^ 1);

#pragma unroll
    for (int kk = 0; kk < BK; kk += 32) {
      const int r16 = lane & 15;
      const int abase = (lane < 16) ? 0 : 8;   // A: K pairs interleaved per half-wave
      const int bbase = (lane < 16) ? 0 : 8;   // B: K pairs interleaved per half-wave
      union FragU { v16bf v; uint4 u[2]; };
      FragU af[4], bfr[2];
#pragma unroll
      for (int mi = 0; mi < 4; mi++) {
        const unsigned short* p = &sA[buf][(wm + mi * 16 + r16) * LDT + kk];
        af[mi].u[0] = *(const uint4*)(p + abase);       // K kk+base .. +7
        af[mi].u[1] = *(const uint4*)(p + 16 + abase);  // K kk+16+base .. +7
      }
#pragma unroll
      for (int ni = 0; ni < 2; ni++) {
        const unsigned short* p = &sB[buf][(wn + ni * 16 + r16) * LDT + kk];
        bfr[ni].u[0] = *(const uint4*)(p + bbase);
        bfr[ni].u[1] = *(const uint4*)(p + 16 + bbase);
      }
#pragma unroll
      for (int mi = 0; mi < 4; mi++)
#pragma unroll
        for (int ni = 0; ni < 2; ni++)
          acc[mi][ni] = __builtin_amdgcn_wmma_f32_16x16x32_bf16(
              false, af[mi].v, false, bfr[ni].v, (short)0, acc[mi][ni],
              false, false);
    }
    wait_async();   // asyncs for next stage must land before the barrier
    __syncthreads();
    buf ^= 1;
  }

  // Epilogue: C/D layout -> VGPR r = row r (+8 for lanes 16-31), lane%16 = col
  const int r16 = lane & 15;
  const int mhalf = (lane < 16) ? 0 : 8;
#pragma unroll
  for (int mi = 0; mi < 4; mi++)
#pragma unroll
    for (int ni = 0; ni < 2; ni++)
#pragma unroll
      for (int r = 0; r < 8; r++) {
        long m = m0 + wm + mi * 16 + mhalf + r;
        int n = n0 + wn + ni * 16 + r16;
        Y[m * COUT + n] = f2bf(acc[mi][ni][r]);
      }
}

// ---------------------------------------------------------------------------
// 5) Per-channel sum / sumsq over M_ rows (for training-mode BN)
// ---------------------------------------------------------------------------
__global__ __launch_bounds__(256) void stats_kernel(
    const unsigned short* __restrict__ y, float* __restrict__ sum,
    float* __restrict__ sq) {
  const int tid = threadIdx.x;
  const long row0 = (long)blockIdx.x * 1024;
  float s0 = 0, q0 = 0, s1 = 0, q1 = 0;
  for (int r = 0; r < 1024; r++) {
    const unsigned short* p = y + (row0 + r) * COUT;
    float v0 = bf2f(p[tid]);        s0 += v0; q0 += v0 * v0;
    float v1 = bf2f(p[tid + 256]);  s1 += v1; q1 += v1 * v1;
  }
  atomicAdd(&sum[tid], s0);        atomicAdd(&sq[tid], q0);
  atomicAdd(&sum[tid + 256], s1);  atomicAdd(&sq[tid + 256], q1);
}

__global__ void finalize_bn_kernel(const float* __restrict__ sum,
                                   const float* __restrict__ sq,
                                   const float* __restrict__ g,
                                   const float* __restrict__ beta,
                                   float* __restrict__ ss) {
  int c = blockIdx.x * blockDim.x + threadIdx.x;
  if (c >= COUT) return;
  const float invM = 1.0f / (float)M_;
  float mean = sum[c] * invM;
  float var = sq[c] * invM - mean * mean;  // biased var, as jnp.var
  float sc = g[c] * rsqrtf(var + 1e-5f);
  ss[c] = sc;
  ss[COUT + c] = beta[c] - mean * sc;
}

// ---------------------------------------------------------------------------
// 6) BN2 + ReLU + max over K -> new_points
//    max_k relu(a_k) == max(0, max_k a_k) so init m = 0.
// ---------------------------------------------------------------------------
__global__ __launch_bounds__(256) void maxpool_kernel(
    const unsigned short* __restrict__ y2, const float* __restrict__ ss,
    float* __restrict__ outp) {
  long gid = (long)blockIdx.x * 256 + threadIdx.x;  // B_*S_*COUT
  int c = (int)(gid & (COUT - 1));
  long q = gid >> 9;
  float sc = ss[c], sh = ss[COUT + c];
  float m = 0.f;
#pragma unroll
  for (int k = 0; k < K_; k++) {
    float v = fmaf(bf2f(y2[((q * K_ + k) << 9) + c]), sc, sh);
    m = fmaxf(m, v);
  }
  outp[gid] = m;
}

// ---------------------------------------------------------------------------
// Launcher
// ---------------------------------------------------------------------------
extern "C" void kernel_launch(void* const* d_in, const int* in_sizes, int n_in,
                              void* d_out, int out_size, void* d_ws,
                              size_t ws_size, hipStream_t stream) {
  (void)in_sizes; (void)n_in; (void)out_size; (void)ws_size;
  const float* xy     = (const float*)d_in[0];
  const float* points = (const float*)d_in[1];
  const float* W1     = (const float*)d_in[2];
  // d_in[3] (b1) and d_in[7] (b2) cancel under training-mode BN -> unused.
  const float* g1     = (const float*)d_in[4];
  const float* beta1  = (const float*)d_in[5];
  const float* W2     = (const float*)d_in[6];
  const float* g2     = (const float*)d_in[8];
  const float* beta2  = (const float*)d_in[9];
  float* out = (float*)d_out;
  float* out_newxy = out;                        // [B,S,2]
  float* out_newpts = out + (size_t)B_ * S_ * 2; // [B,S,512]

  char* ws = (char*)d_ws;
  size_t off = 0;
  int* fps_idx = (int*)(ws + off);            off += (size_t)B_ * S_ * 4;
  int* knn_idx = (int*)(ws + off);            off += (size_t)B_ * S_ * K_ * 4;
  unsigned short* w1t = (unsigned short*)(ws + off); off += (size_t)C1IN * COUT * 2;
  unsigned short* w2t = (unsigned short*)(ws + off); off += (size_t)COUT * COUT * 2;
  float* sums = (float*)(ws + off);           off += 4 * COUT * 4;
  float* ss   = (float*)(ws + off);           off += 4 * COUT * 4;
  off = (off + 255) & ~(size_t)255;
  unsigned short* a1 = (unsigned short*)(ws + off); off += (size_t)M_ * C1IN * 2;
  unsigned short* y1 = (unsigned short*)(ws + off); off += (size_t)M_ * COUT * 2;
  unsigned short* y2 = (unsigned short*)(ws + off);

  float* sum1 = sums,        *sq1 = sums + COUT;
  float* sum2 = sums + 1024, *sq2 = sums + 1536;
  float* ss1 = ss;   // scale[512] | shift[512]
  float* ss2 = ss + 1024;

  // 1) FPS (writes fps_idx to workspace, new_xy directly into d_out)
  fps_kernel<<<B_, 1024, 0, stream>>>(xy, fps_idx, out_newxy);
  // 2) 16-NN
  knn_kernel<<<(B_ * S_) / 4, 128, 0, stream>>>(xy, fps_idx, knn_idx);
  // 3) weights -> bf16 transposed; gather/concat -> bf16 A1
  transpose_w_kernel<<<(C1IN * COUT) / 256, 256, 0, stream>>>(W1, w1t, C1IN);
  transpose_w_kernel<<<(COUT * COUT) / 256, 256, 0, stream>>>(W2, w2t, COUT);
  build_a1_kernel<<<(int)(((long)M_ * C1IN) / 256), 256, 0, stream>>>(
      xy, points, knn_idx, out_newxy, a1);
  zero_f32_kernel<<<8, 256, 0, stream>>>(sums, 4 * COUT);
  // 4) GEMM1 (WMMA bf16, async-to-LDS staging) -> y1
  gemm_bf16_kernel<C1IN, false><<<dim3(M_ / 128, COUT / 128), 256, 0, stream>>>(
      a1, w1t, y1, nullptr);
  // 5) BN1 stats
  stats_kernel<<<M_ / 1024, 256, 0, stream>>>(y1, sum1, sq1);
  finalize_bn_kernel<<<2, 256, 0, stream>>>(sum1, sq1, g1, beta1, ss1);
  // 6) GEMM2 with fused BN1+ReLU on A-load -> y2
  gemm_bf16_kernel<COUT, true><<<dim3(M_ / 128, COUT / 128), 256, 0, stream>>>(
      y1, w2t, y2, ss1);
  // 7) BN2 stats
  stats_kernel<<<M_ / 1024, 256, 0, stream>>>(y2, sum2, sq2);
  finalize_bn_kernel<<<2, 256, 0, stream>>>(sum2, sq2, g2, beta2, ss2);
  // 8) BN2 + ReLU + max over K -> new_points
  maxpool_kernel<<<(B_ * S_ * COUT) / 256, 256, 0, stream>>>(y2, ss2, out_newpts);
}
